// ContextPulseAttention_23862838296824
// MI455X (gfx1250) — matrix operand order repaired and checked by reading, not compiled, not run on hardware
//
#include <hip/hip_runtime.h>

// ---------------------------------------------------------------------------
// Problem constants (from the reference)
// ---------------------------------------------------------------------------
#define BB     2
#define TT     2048
#define DMODEL 1024
#define NHEAD  16
#define DH     64
#define CHUNK  16
#define NCHUNK (TT / CHUNK)          // 128
#define ROWS   (BB * TT)             // 4096 token rows
#define BH     (BB * NHEAD)          // 32
#define KD     1024                  // K-dim of every dense GEMM here

typedef unsigned short ushort_t;
typedef unsigned int   uint_t;
typedef float    v8f   __attribute__((ext_vector_type(8)));
typedef __bf16   v16bf __attribute__((ext_vector_type(16)));
typedef uint_t   v4u   __attribute__((ext_vector_type(4)));
typedef int      vsi4  __attribute__((vector_size(16)));   // matches builtin param

union Frag16 { v16bf v; ushort_t u[16]; uint_t w[8]; };

// ---------------------------------------------------------------------------
// Async global->LDS copy (CDNA5 path, guarded) with synchronous fallback
// ---------------------------------------------------------------------------
#if __has_builtin(__builtin_amdgcn_global_load_async_to_lds_b128)
__device__ __forceinline__ void async_copy16(const void* gsrc, void* ldst) {
    __builtin_amdgcn_global_load_async_to_lds_b128((vsi4*)gsrc, (vsi4*)ldst, 0, 0);
}
__device__ __forceinline__ void async_wait0() {
#if __has_builtin(__builtin_amdgcn_s_wait_asynccnt)
    __builtin_amdgcn_s_wait_asynccnt(0);
#else
    asm volatile("s_wait_asynccnt 0" ::: "memory");
#endif
}
#else
__device__ __forceinline__ void async_copy16(const void* gsrc, void* ldst) {
    *(v4u*)ldst = *(const v4u*)gsrc;
}
__device__ __forceinline__ void async_wait0() {}
#endif

__device__ __forceinline__ ushort_t f2bf(float f) {
    uint_t u = __builtin_bit_cast(uint_t, f);
    return (ushort_t)((u + 0x7FFFu + ((u >> 16) & 1u)) >> 16);
}

// A-fragment K map (16-bit A 16x32, contiguous: lanes 0-15 -> K 0..15, lanes 16-31 -> K 16..31)
__device__ __forceinline__ int kmapA(int lane, int h) {
    return ((lane >> 4) << 4) + h;
}
// B-fragment K map (16-bit B 32x16: lanes 0-15 -> K 0..15, lanes 16-31 -> K 16..31)
__device__ __forceinline__ int kmapB(int lane, int h) {
    return ((lane >> 4) << 4) + h;
}

// Contiguous bf16 A-fragment from a row-major row (pairs are K-consecutive)
__device__ __forceinline__ v16bf load_fragA_rm(const ushort_t* rowptr, int kbase, int lane) {
    Frag16 f;
    const int hi16 = (lane >> 4) << 4;
#pragma unroll
    for (int j = 0; j < 8; ++j) {
        int koff = kbase + hi16 + (j << 1);
        f.w[j] = *(const uint_t*)(rowptr + koff);
    }
    return f.v;
}
// Contiguous bf16 B-fragment from a row-major row (fully K-consecutive)
__device__ __forceinline__ v16bf load_fragB_rm(const ushort_t* rowptr, int kbase, int lane) {
    Frag16 f;
    const uint_t* p = (const uint_t*)(rowptr + kbase + ((lane >> 4) << 4));
#pragma unroll
    for (int j = 0; j < 8; ++j) f.w[j] = p[j];
    return f.v;
}
// A-fragment gathered from an fp32 row (convert on the fly)
__device__ __forceinline__ v16bf load_fragA_f32(const float* rowptr, int kbase, int lane) {
    Frag16 f;
#pragma unroll
    for (int h = 0; h < 16; ++h) f.u[h] = f2bf(rowptr[kbase + kmapA(lane, h)]);
    return f.v;
}

__device__ __forceinline__ v8f wmma_bf16(v16bf a, v16bf b, v8f c) {
    return __builtin_amdgcn_wmma_f32_16x16x32_bf16(false, a, false, b, (short)0, c, false, false);
}

// Alpert basis columns (L=16, K=2), closed form
__device__ __forceinline__ float wb0() { return 0.25f; }
__device__ __forceinline__ float wb1(int l) { return (float)(2 * l - 15) * (1.0f / 36.878178f); }

// ---------------------------------------------------------------------------
// fp32 -> bf16 convert
// ---------------------------------------------------------------------------
__global__ __launch_bounds__(256) void cvt_f32_bf16(const float* __restrict__ in,
                                                    ushort_t* __restrict__ out, int n) {
    int i = blockIdx.x * 256 + threadIdx.x;
    if (i < n) out[i] = f2bf(in[i]);
}

// ---------------------------------------------------------------------------
// Generic WMMA GEMM: C[m,n] = sum_k A[m,k] * W[n,k]   (A: MxKD, W: NxKD, bf16)
// All 8 waves of a block share one 16-row A tile (ntiles % 8 == 0), staged
// once into LDS via async global->LDS; W rows stream from global (prefetched).
// mode 0: fp32 row-major MxN              (final Wo projection -> d_out)
// mode 1: fp32 to (b,h,t,dh) layout       (q_raw / k_raw)
// mode 2: sigmoid(y + bias[n]) -> (b,h,t,dh) fp32 (gate)
// mode 3: bf16 to (b,h,t,dh) layout       (v)
// ---------------------------------------------------------------------------
__global__ __launch_bounds__(256) void gemm_abt_bf16(const ushort_t* __restrict__ Abf,
                                                     const ushort_t* __restrict__ Wbf,
                                                     const float* __restrict__ bias,
                                                     void* __restrict__ dst,
                                                     int M, int N, int mode) {
    __shared__ ushort_t ldsA[16 * KD];               // 32 KB, block-shared A tile
    const int tid  = threadIdx.x;
    const int lane = tid & 31;
    const int wid  = tid >> 5;
    const int ntiles = N >> 4;
    const int gw0 = blockIdx.x * 8;
    const int mt  = gw0 / ntiles;                    // block-uniform (8 | ntiles)

    // Stage A tile: rows mt*16..mt*16+15 are one contiguous 32 KB span.
    {
        const char* src = (const char*)(Abf + (size_t)mt * 16 * KD);
        char*       dstl = (char*)ldsA;
        const int byteoff = tid * 128;               // 256 threads x 128 B
#pragma unroll
        for (int j = 0; j < 8; ++j)
            async_copy16(src + byteoff + j * 16, dstl + byteoff + j * 16);
    }
    async_wait0();
    __syncthreads();

    const int gw = gw0 + wid;
    if (gw >= (M >> 4) * ntiles) return;
    const int nt = gw - mt * ntiles;

    const ushort_t* arow = ldsA + (size_t)(lane & 15) * KD;
    const ushort_t* brow = Wbf + (size_t)(nt * 16 + (lane & 15)) * KD;

    v8f acc = {};
    for (int kb = 0; kb < KD; kb += 32) {
        __builtin_prefetch(brow + kb + 64, 0, 1);    // global_prefetch_b8
        v16bf a = load_fragA_rm(arow, kb, lane);     // ds loads
        v16bf b = load_fragB_rm(brow, kb, lane);     // global b128 clauses
        acc = wmma_bf16(a, b, acc);
    }

    const int colg = nt * 16 + (lane & 15);
#pragma unroll
    for (int v = 0; v < 8; ++v) {
        int rowg = mt * 16 + v + ((lane >> 4) << 3);
        float y = acc[v];
        if (mode == 0) {
            ((float*)dst)[(size_t)rowg * N + colg] = y;
        } else {
            int b_ = rowg >> 11, t_ = rowg & (TT - 1);
            int h_ = colg >> 6,  d_ = colg & (DH - 1);
            size_t idx = (((size_t)(b_ * NHEAD + h_) * TT) + t_) * DH + d_;
            if (mode == 1)      ((float*)dst)[idx] = y;
            else if (mode == 2) { float z = y + bias[colg];
                                  ((float*)dst)[idx] = 1.f / (1.f + __expf(-z)); }
            else                ((ushort_t*)dst)[idx] = f2bf(y);
        }
    }
}

// ---------------------------------------------------------------------------
// Joint q/k normalization (mean/std over head dim, ddof=1) + RoPE, in place.
// One wave per (b,h,t) row; lane e handles pair (2e,2e+1) -> outputs (e, e+32).
// ---------------------------------------------------------------------------
__global__ __launch_bounds__(256) void norm_rope(float* __restrict__ q, float* __restrict__ k) {
    const int lane = threadIdx.x & 31;
    const int wid  = threadIdx.x >> 5;
    const int row  = blockIdx.x * 8 + wid;          // 0 .. BH*TT-1
    float* qr = q + (size_t)row * DH;
    float* kr = k + (size_t)row * DH;

    float q1 = qr[2 * lane], q2 = qr[2 * lane + 1];
    float k1 = kr[2 * lane], k2 = kr[2 * lane + 1];

    float sq = q1 + q2, sqq = q1 * q1 + q2 * q2;
    float sk = k1 + k2, skk = k1 * k1 + k2 * k2;
#pragma unroll
    for (int m = 16; m >= 1; m >>= 1) {
        sq  += __shfl_xor(sq,  m, 32);
        sqq += __shfl_xor(sqq, m, 32);
        sk  += __shfl_xor(sk,  m, 32);
        skk += __shfl_xor(skk, m, 32);
    }
    float mq = sq * (1.f / 64.f), mk = sk * (1.f / 64.f);
    float vq = (sqq - 64.f * mq * mq) * (1.f / 63.f);
    float vk = (skk - 64.f * mk * mk) * (1.f / 63.f);
    float ms  = 0.5f * (mq + mk);
    float ss  = 0.5f * (sqrtf(fmaxf(vq, 0.f)) + sqrtf(fmaxf(vk, 0.f)));
    float inv = 1.f / (ss + 1e-6f);
    q1 = (q1 - ms) * inv; q2 = (q2 - ms) * inv;
    k1 = (k1 - ms) * inv; k2 = (k2 - ms) * inv;

    int t = row & (TT - 1);
    // inv_freq[e] = 10000^{-e/32}
    float ang = (float)t * __expf(-(float)lane * (9.210340372f / 32.f));
    float sn, cs; __sincosf(ang, &sn, &cs);

    qr[lane]      = q1 * cs - q2 * sn;
    qr[lane + 32] = q1 * sn + q2 * cs;
    kr[lane]      = k1 * cs - k2 * sn;
    kr[lane + 32] = k1 * sn + k2 * cs;
}

// ---------------------------------------------------------------------------
// Wedge: z <- z @ (I + A_h - A_h^T), in place, via WMMA. One wave per 16-row
// tile of one (b,h). Optionally writes a bf16 copy (k_geo for attention).
// ---------------------------------------------------------------------------
__global__ __launch_bounds__(256) void wedge(float* __restrict__ z, ushort_t* __restrict__ zbf,
                                             const float* __restrict__ A) {
    const int lane = threadIdx.x & 31;
    const int wid  = threadIdx.x >> 5;
    const int gw   = blockIdx.x * 8 + wid;          // 0..BH*128-1
    const int bh   = gw >> 7;
    const int mt   = gw & 127;
    const int head = bh & (NHEAD - 1);
    const float* Ah = A + (size_t)head * DH * DH;
    float* zb = z + ((size_t)bh * TT + mt * 16) * DH;

    const float* arow = zb + (size_t)(lane & 15) * DH;
    v16bf a0 = load_fragA_f32(arow, 0, lane);
    v16bf a1 = load_fragA_f32(arow, 32, lane);

#pragma unroll
    for (int nt = 0; nt < 4; ++nt) {
        v8f acc = {};
        const int e = nt * 16 + (lane & 15);
#pragma unroll
        for (int kf = 0; kf < 2; ++kf) {
            Frag16 bfr;
#pragma unroll
            for (int h = 0; h < 16; ++h) {
                int d = kf * 32 + kmapB(lane, h);
                float m = ((d == e) ? 1.f : 0.f) + Ah[d * DH + e] - Ah[e * DH + d];
                bfr.u[h] = f2bf(m);
            }
            acc = wmma_bf16(kf == 0 ? a0 : a1, bfr.v, acc);
        }
#pragma unroll
        for (int v = 0; v < 8; ++v) {
            int r = v + ((lane >> 4) << 3);
            size_t idx = (size_t)r * DH + e;
            zb[idx] = acc[v];
            if (zbf) zbf[((size_t)bh * TT + mt * 16) * DH + idx] = f2bf(acc[v]);
        }
    }
}

// ---------------------------------------------------------------------------
// Sequential chunk scan. One block per (b,h); 128 steps; history (128 x 2 x 64
// fp32 = 64 KB) lives in dynamic LDS. Emits Q in bf16 (b,h,t,d).
// ---------------------------------------------------------------------------
__global__ __launch_bounds__(256) void scan_kernel(const float* __restrict__ qg,
                                                   const float* __restrict__ kg,
                                                   const float* __restrict__ gate,
                                                   ushort_t* __restrict__ Qbf,
                                                   const float* __restrict__ jump_scale,
                                                   const float* __restrict__ jump_bias) {
    extern __shared__ float hist[];                 // NCHUNK * 128 floats
    __shared__ float kcoef[128], retr[128], sc[128], wgt[128];
    __shared__ float qout[CHUNK * DH], lastv[DH];
    __shared__ float red[256];
    __shared__ float sh_jg;

    const int tid = threadIdx.x;
    const int bh  = blockIdx.x;
    const size_t base = (size_t)bh * TT * DH;
    const float js = jump_scale[0], jb = jump_bias[0];

    if (tid < DH) lastv[tid] = 0.f;
    __syncthreads();

    for (int i = 0; i < NCHUNK; ++i) {
        const size_t cbase = base + (size_t)i * CHUNK * DH;

        // k_coef[k,d] = sum_l k_c[l,d] * Wb[l,k]
        if (tid < 128) {
            int kk = tid >> 6, d = tid & 63;
            float s = 0.f;
#pragma unroll
            for (int l = 0; l < CHUNK; ++l)
                s += kg[cbase + l * DH + d] * (kk == 0 ? wb0() : wb1(l));
            kcoef[tid] = s;
        }
        __syncthreads();

        // scores vs history
        if (tid < 128) {
            float s = -1e30f;
            if (tid < i) {
                float acc = 0.f;
                for (int kd = 0; kd < 128; ++kd) acc += kcoef[kd] * hist[tid * 128 + kd];
                s = acc * 0.125f;                    // SCALE = 64^-0.5
            }
            sc[tid] = s;
        }
        __syncthreads();

        // max over 128
        red[tid] = (tid < 128) ? sc[tid] : -1e30f;
        __syncthreads();
        for (int st = 128; st > 0; st >>= 1) {
            if (tid < st) red[tid] = fmaxf(red[tid], red[tid + st]);
            __syncthreads();
        }
        const float maxsc = red[0];
        if (tid == 0) sh_jg = 1.f / (1.f + __expf(-(maxsc * js + jb)));
        __syncthreads();

        // softmax weights (unnormalized) + sum
        float wv = (tid < 128 && tid < i) ? __expf(sc[tid] - maxsc) : 0.f;
        if (tid < 128) wgt[tid] = wv;
        red[tid] = wv;
        __syncthreads();
        for (int st = 128; st > 0; st >>= 1) {
            if (tid < st) red[tid] += red[tid + st];
            __syncthreads();
        }
        const float wsum = red[0] + 1e-30f;

        // retrieval (normalized)
        if (tid < 128) {
            float s = 0.f;
            for (int n = 0; n < i; ++n) s += wgt[n] * hist[n * 128 + tid];
            retr[tid] = s / wsum;
        }
        __syncthreads();

        // q_out = cumsum_l(g*q) + inj ; write bf16 Q
        if (tid < DH) {
            const int d = tid;
            const float jg = sh_jg;
            const float r0 = retr[d], r1 = retr[64 + d];
            const float lst = lastv[d];
            float cum = 0.f;
#pragma unroll
            for (int l = 0; l < CHUNK; ++l) {
                float qrec = r0 * wb0() + r1 * wb1(l);
                float inj  = (i > 0) ? ((1.f - jg) * lst + jg * qrec) : 0.f;
                cum += gate[cbase + l * DH + d] * qg[cbase + l * DH + d];
                float qo = cum + inj;
                qout[l * DH + d] = qo;
                Qbf[cbase + l * DH + d] = f2bf(qo);
            }
            lastv[d] = qout[15 * DH + d];
        }
        __syncthreads();

        // hist[i,k,d] = sum_l q_out[l,d] * Wb[l,k]
        if (tid < 128) {
            int kk = tid >> 6, d = tid & 63;
            float s = 0.f;
#pragma unroll
            for (int l = 0; l < CHUNK; ++l)
                s += qout[l * DH + d] * (kk == 0 ? wb0() : wb1(l));
            hist[i * 128 + tid] = s;
        }
        __syncthreads();
    }
}

// ---------------------------------------------------------------------------
// Flash attention with sink column + v_null. One wave per (b,h, 16-row qtile).
// Per 32-key step: async-stage the 32x64 V tile to per-wave LDS (overlaps the
// QK^T WMMAs + softmax), 2+2 S-WMMAs, online softmax (shuffle reductions),
// P staged per-wave through LDS (s_wait_dscnt ordering), 4 PV-WMMAs.
// No __syncthreads anywhere (key-loop trip count diverges across waves);
// ASYNCcnt/DScnt are per-wave so per-wave waits suffice.
// ---------------------------------------------------------------------------
__global__ __launch_bounds__(256) void flash_attn(const ushort_t* __restrict__ Qbf,
                                                  const ushort_t* __restrict__ Kbf,
                                                  const ushort_t* __restrict__ Vbf,
                                                  const float* __restrict__ sink_scalar,
                                                  const float* __restrict__ v_null,
                                                  ushort_t* __restrict__ outbf) {
    __shared__ ushort_t p_lds[8][CHUNK * 32];        //  8 KB: P tiles
    __shared__ ushort_t vstage[8][32 * DH];          // 32 KB: V tiles
    const int lane = threadIdx.x & 31;
    const int wid  = threadIdx.x >> 5;
    const int gw   = blockIdx.x * 8 + wid;
    const int bh   = gw >> 7;
    const int qi   = gw & 127;
    const int head = bh & (NHEAD - 1);
    const int b    = bh >> 4;
    const size_t kvbase = (size_t)bh * TT * DH;

    const ushort_t* qrow = Qbf + kvbase + (size_t)(qi * 16 + (lane & 15)) * DH;
    v16bf aq0 = load_fragA_rm(qrow, 0, lane);
    v16bf aq1 = load_fragA_rm(qrow, 32, lane);

    float mrow[8], lrow[8];
    v8f acc[4] = {};
#pragma unroll
    for (int v = 0; v < 8; ++v) { mrow[v] = -1e30f; lrow[v] = 0.f; }

    ushort_t* pl = p_lds[wid];
    ushort_t* vs = vstage[wid];
    const int nsteps = (qi >> 1) + 1;
    const int rhi = (lane >> 4) << 3;

    for (int kb = 0; kb < nsteps; ++kb) {
        const int ks0 = kb * 32;

        // Kick off async V-tile stage: lane copies key row ks0+lane (128 B).
        {
            const char* vsrc = (const char*)(Vbf + kvbase + (size_t)(ks0 + lane) * DH);
            char*       vdst = (char*)(vs + (size_t)lane * DH);
#pragma unroll
            for (int j = 0; j < 8; ++j)
                async_copy16(vsrc + j * 16, vdst + j * 16);
        }

        // S = Q K^T (two 16x16 tiles)
        v8f s0 = {}, s1 = {};
        {
            const ushort_t* k0 = Kbf + kvbase + (size_t)(ks0 + (lane & 15)) * DH;
            const ushort_t* k1 = k0 + 16 * DH;
            s0 = wmma_bf16(aq0, load_fragB_rm(k0, 0, lane), s0);
            s0 = wmma_bf16(aq1, load_fragB_rm(k0, 32, lane), s0);
            s1 = wmma_bf16(aq0, load_fragB_rm(k1, 0, lane), s1);
            s1 = wmma_bf16(aq1, load_fragB_rm(k1, 32, lane), s1);
        }
        const int col0 = ks0 + (lane & 15);
#pragma unroll
        for (int v = 0; v < 8; ++v) {
            const int rowt = qi * 16 + v + rhi;
            float e0 = s0[v] * 0.125f; if (col0 > rowt)      e0 = -1e30f;
            float e1 = s1[v] * 0.125f; if (col0 + 16 > rowt) e1 = -1e30f;
            float tm = fmaxf(e0, e1);
#pragma unroll
            for (int m = 8; m >= 1; m >>= 1) tm = fmaxf(tm, __shfl_xor(tm, m, 32));
            const float mn   = fmaxf(mrow[v], tm);
            const float corr = __expf(mrow[v] - mn);
            const float p0 = __expf(e0 - mn), p1 = __expf(e1 - mn);
            float rs = p0 + p1;
#pragma unroll
            for (int m = 8; m >= 1; m >>= 1) rs += __shfl_xor(rs, m, 32);
            lrow[v] = lrow[v] * corr + rs;
            mrow[v] = mn;
#pragma unroll
            for (int nt = 0; nt < 4; ++nt) acc[nt][v] *= corr;
            const int r = v + rhi;
            pl[r * 32 + (lane & 15)]      = f2bf(p0);
            pl[r * 32 + 16 + (lane & 15)] = f2bf(p1);
        }
        asm volatile("s_wait_dscnt 0" ::: "memory");   // order per-wave LDS RAW (P)
        async_wait0();                                 // V tile resident in LDS
        v16bf ap = load_fragA_rm(pl + (size_t)(lane & 15) * 32, 0, lane);
#pragma unroll
        for (int nt = 0; nt < 4; ++nt) {
            Frag16 bv;
            const int e = nt * 16 + (lane & 15);
#pragma unroll
            for (int h = 0; h < 16; ++h) {
                int keyl = kmapB(lane, h);             // local key index 0..31
                bv.u[h] = vs[(size_t)keyl * DH + e];   // ds_load_u16
            }
            acc[nt] = wmma_bf16(ap, bv.v, acc[nt]);
        }
    }

    const float snk = sink_scalar[head];
    const float* vn = v_null + head * DH;
    const size_t obase = (size_t)b * TT * DMODEL + (size_t)head * DH;
#pragma unroll
    for (int v = 0; v < 8; ++v) {
        const float es = __expf(snk - mrow[v]);
        const float dn = 1.f / (lrow[v] + es);
        const int rowt = qi * 16 + v + rhi;
#pragma unroll
        for (int nt = 0; nt < 4; ++nt) {
            const int e = nt * 16 + (lane & 15);
            float o = acc[nt][v] * dn + es * dn * vn[e];
            outbf[obase + (size_t)rowt * DMODEL + e] = f2bf(o);
        }
    }
}

// ---------------------------------------------------------------------------
// Host-side launch
// ---------------------------------------------------------------------------
extern "C" void kernel_launch(void* const* d_in, const int* in_sizes, int n_in,
                              void* d_out, int out_size, void* d_ws, size_t ws_size,
                              hipStream_t stream) {
    (void)in_sizes; (void)n_in; (void)out_size; (void)ws_size;
    const float* x      = (const float*)d_in[0];
    const float* Wq     = (const float*)d_in[1];
    const float* Wk     = (const float*)d_in[2];
    const float* Wv     = (const float*)d_in[3];
    const float* Wo     = (const float*)d_in[4];
    const float* gate_w = (const float*)d_in[5];
    const float* gate_b = (const float*)d_in[6];
    const float* A      = (const float*)d_in[7];
    const float* jsc    = (const float*)d_in[8];
    const float* jbi    = (const float*)d_in[9];
    const float* sink   = (const float*)d_in[10];
    const float* vnull  = (const float*)d_in[11];

    size_t off = 0;
    char* wsb = (char*)d_ws;
    auto take = [&](size_t bytes) -> void* {
        void* p = wsb + off;
        off += (bytes + 255) & ~(size_t)255;
        return p;
    };
    const size_t SZ_BHTD_F = (size_t)BH * TT * DH * sizeof(float);     // 16 MB
    const size_t SZ_BHTD_H = (size_t)BH * TT * DH * sizeof(ushort_t);  //  8 MB
    const size_t SZ_MAT_H  = (size_t)DMODEL * DMODEL * sizeof(ushort_t);
    const size_t SZ_X_H    = (size_t)ROWS * DMODEL * sizeof(ushort_t);

    float*    qbuf   = (float*)take(SZ_BHTD_F);   // q_raw -> q_rot -> q_geo
    float*    kbuf   = (float*)take(SZ_BHTD_F);   // k_raw -> k_rot -> k_geo
    float*    gbuf   = (float*)take(SZ_BHTD_F);   // gate (post-sigmoid)
    ushort_t* xbf    = (ushort_t*)take(SZ_X_H);
    ushort_t* Wqbf   = (ushort_t*)take(SZ_MAT_H);
    ushort_t* Wkbf   = (ushort_t*)take(SZ_MAT_H);
    ushort_t* Wvbf   = (ushort_t*)take(SZ_MAT_H);
    ushort_t* Wgbf   = (ushort_t*)take(SZ_MAT_H);
    ushort_t* Wobf   = (ushort_t*)take(SZ_MAT_H);
    ushort_t* vbf    = (ushort_t*)take(SZ_BHTD_H);
    ushort_t* kgbf   = (ushort_t*)take(SZ_BHTD_H);
    ushort_t* Qbf    = (ushort_t*)take(SZ_BHTD_H);
    ushort_t* attnbf = (ushort_t*)take(SZ_X_H);

    const int NX = ROWS * DMODEL;        // 4M
    const int NW = DMODEL * DMODEL;      // 1M
    cvt_f32_bf16<<<(NX + 255) / 256, 256, 0, stream>>>(x, xbf, NX);
    cvt_f32_bf16<<<(NW + 255) / 256, 256, 0, stream>>>(Wq, Wqbf, NW);
    cvt_f32_bf16<<<(NW + 255) / 256, 256, 0, stream>>>(Wk, Wkbf, NW);
    cvt_f32_bf16<<<(NW + 255) / 256, 256, 0, stream>>>(Wv, Wvbf, NW);
    cvt_f32_bf16<<<(NW + 255) / 256, 256, 0, stream>>>(gate_w, Wgbf, NW);
    cvt_f32_bf16<<<(NW + 255) / 256, 256, 0, stream>>>(Wo, Wobf, NW);

    const int gemm_blocks = (ROWS / 16) * (DMODEL / 16) / 8;   // 2048
    gemm_abt_bf16<<<gemm_blocks, 256, 0, stream>>>(xbf, Wqbf, nullptr, qbuf, ROWS, DMODEL, 1);
    gemm_abt_bf16<<<gemm_blocks, 256, 0, stream>>>(xbf, Wkbf, nullptr, kbuf, ROWS, DMODEL, 1);
    gemm_abt_bf16<<<gemm_blocks, 256, 0, stream>>>(xbf, Wvbf, nullptr, vbf,  ROWS, DMODEL, 3);
    gemm_abt_bf16<<<gemm_blocks, 256, 0, stream>>>(xbf, Wgbf, gate_b,  gbuf, ROWS, DMODEL, 2);

    norm_rope<<<(BH * TT) / 8, 256, 0, stream>>>(qbuf, kbuf);

    wedge<<<(BH * 128) / 8, 256, 0, stream>>>(qbuf, nullptr, A);
    wedge<<<(BH * 128) / 8, 256, 0, stream>>>(kbuf, kgbf,    A);

    scan_kernel<<<BH, 256, NCHUNK * 128 * sizeof(float), stream>>>(qbuf, kbuf, gbuf, Qbf, jsc, jbi);

    flash_attn<<<(BH * 128) / 8, 256, 0, stream>>>(Qbf, kgbf, vbf, sink, vnull, attnbf);

    gemm_abt_bf16<<<gemm_blocks, 256, 0, stream>>>(attnbf, Wobf, nullptr, d_out, ROWS, DMODEL, 0);
}
